// MoELayer_5566277616585
// MI455X (gfx1250) — matrix operand order repaired and checked by reading, not compiled, not run on hardware
//
#include <hip/hip_runtime.h>
#include <math.h>

// ---------------- CDNA5 WMMA types ----------------
typedef __attribute__((ext_vector_type(16))) __bf16 v16bf;
typedef __attribute__((ext_vector_type(8)))  __bf16 v8bf;
typedef __attribute__((ext_vector_type(8)))  float  v8f;

#define BM 128
#define BN 128
#define BK 32
#define LDS_PAD 40   // 40 bf16 = 80B row stride; 8-elem (16B) chunks stay 16B-aligned

__device__ __forceinline__ float selu_f(float x) {
    const float scale = 1.0507009873554804934193349852946f;
    const float alpha = 1.6732632423543772848170429916717f;
    return x > 0.f ? scale * x : scale * alpha * (expf(x) - 1.f);
}

// Stage one 128x32 A tile and one 32x128 B tile (transposed) into LDS as bf16.
// LDA/LDW compile-time -> B's k-strided loads become clause-able imm-offset loads.
template<int LDA, int LDW>
__device__ __forceinline__ void stage_tile(
    const float* __restrict__ A, const float* __restrict__ W,
    int m0, int n0, int k0, int tid,
    __bf16 (*As)[LDS_PAD], __bf16 (*Bs)[LDS_PAD])
{
    #pragma unroll
    for (int i = 0; i < 2; ++i) {
        int idx8 = tid + i * 256;            // 0..511 octets
        int r    = idx8 >> 2;                // 0..127
        int cseg = (idx8 & 3) * 8;           // 0,8,16,24
        const float4 f0 = *(const float4*)&A[(size_t)(m0 + r) * LDA + k0 + cseg];
        const float4 f1 = *(const float4*)&A[(size_t)(m0 + r) * LDA + k0 + cseg + 4];
        v8bf v;
        v[0] = (__bf16)f0.x; v[1] = (__bf16)f0.y; v[2] = (__bf16)f0.z; v[3] = (__bf16)f0.w;
        v[4] = (__bf16)f1.x; v[5] = (__bf16)f1.y; v[6] = (__bf16)f1.z; v[7] = (__bf16)f1.w;
        *(v8bf*)&As[r][cseg] = v;
    }
    #pragma unroll
    for (int i = 0; i < 2; ++i) {
        int idx8 = tid + i * 256;            // 0..511 octets
        int n    = idx8 & 127;               // 0..127 (lane-coalesced along N)
        int koct = (idx8 >> 7) * 8;          // 0,8,16,24
        const float* wp = &W[(size_t)(k0 + koct) * LDW + n0 + n];
        v8bf v;
        #pragma unroll
        for (int j = 0; j < 8; ++j)
            v[j] = (__bf16)wp[j * LDW];      // j*LDW*4 fits imm24 -> s_clause group
        *(v8bf*)&Bs[n][koct] = v;
    }
}

// C = act(A[M,KDIM] @ W[KDIM,Ncols] + bias)      (WEIGHTED==0: store f32)
// C += wcol[m] * act(A@W + bias)                 (WEIGHTED==1: accumulate into f32 C)
// `active` (optional): per-128-row-block skip flag (uniform early exit).
template<int ACT, int WEIGHTED, int LDA, int LDW, int LDC, int KDIM>
__global__ __launch_bounds__(256) void gemm_bf16w(
    const float* __restrict__ A,
    const float* __restrict__ W,
    const float* __restrict__ bias,
    float* __restrict__ C,
    const float* __restrict__ wcol,
    const unsigned char* __restrict__ active)
{
    static_assert(KDIM % BK == 0, "K must be multiple of BK");
    if (active && !active[blockIdx.y]) return;   // uniform: EXEC stays all-ones

    __shared__ __align__(16) __bf16 As[2][BM][LDS_PAD];   // [buf][m][k]
    __shared__ __align__(16) __bf16 Bs[2][BN][LDS_PAD];   // [buf][n][k] (transposed)

    const int tid  = threadIdx.x;
    const int lane = tid & 31;
    const int wave = tid >> 5;
    const int l15  = lane & 15;
    const int kh   = lane >> 4;              // 0: lanes 0-15, 1: lanes 16-31
    const int mbase = (wave & 3) * 32;       // wave's M offset inside block tile
    const int nbase = (wave >> 2) * 64;      // wave's N offset inside block tile
    const int m0   = blockIdx.y * BM;
    const int n0   = blockIdx.x * BN;

    v8f acc[2][4];
    #pragma unroll
    for (int ms = 0; ms < 2; ++ms)
        #pragma unroll
        for (int ns = 0; ns < 4; ++ns)
            acc[ms][ns] = (v8f){};

    // prologue: stage tile 0
    stage_tile<LDA, LDW>(A, W, m0, n0, 0, tid, As[0], Bs[0]);
    __syncthreads();

    int buf = 0;
    for (int k0 = 0; k0 < KDIM; k0 += BK) {
        // issue next tile's global loads before consuming current tile
        if (k0 + BK < KDIM) {
            stage_tile<LDA, LDW>(A, W, m0, n0, k0 + BK, tid, As[buf ^ 1], Bs[buf ^ 1]);
            // distance-2 prefetch of the HBM-resident A stream into L2/WGP$
            if (k0 + 2 * BK < KDIM) {
                int r = tid >> 1;
                __builtin_prefetch(&A[(size_t)(m0 + r) * LDA + k0 + 2 * BK + (tid & 1) * 16], 0, 1);
            }
        }

        // ---- fragments per documented 16-bit wave32 layouts ----
        v16bf a[2];
        #pragma unroll
        for (int ms = 0; ms < 2; ++ms) {
            int m = mbase + ms * 16 + l15;
            #pragma unroll
            for (int j = 0; j < 8; ++j) {
                a[ms][j]     = As[buf][m][kh * 8 + j];
                a[ms][8 + j] = As[buf][m][16 + kh * 8 + j];
            }
        }
        v16bf b[4];
        #pragma unroll
        for (int ns = 0; ns < 4; ++ns) {
            int n = nbase + ns * 16 + l15;
            #pragma unroll
            for (int j = 0; j < 16; ++j)
                b[ns][j] = Bs[buf][n][(j & 8) * 2 + kh * 8 + (j & 7)];
        }
        #pragma unroll
        for (int ms = 0; ms < 2; ++ms)
            #pragma unroll
            for (int ns = 0; ns < 4; ++ns)
                acc[ms][ns] = __builtin_amdgcn_wmma_f32_16x16x32_bf16(
                    false, a[ms], false, b[ns], (short)0, acc[ms][ns], false, false);

        // one barrier per K-step: covers (a) next tile staged before its compute,
        // (b) this tile fully read before its buffer is overwritten next iteration
        __syncthreads();
        buf ^= 1;
    }

    // ---- epilogue: C/D layout -> VGPR r holds M = r + 8*kh, N = l15 ----
    #pragma unroll
    for (int ns = 0; ns < 4; ++ns) {
        const int n_g = n0 + nbase + ns * 16 + l15;
        const float bn = bias[n_g];
        #pragma unroll
        for (int ms = 0; ms < 2; ++ms) {
            #pragma unroll
            for (int r = 0; r < 8; ++r) {
                int m = m0 + mbase + ms * 16 + r + 8 * kh;
                float v = acc[ms][ns][r] + bn;
                if (ACT) v = selu_f(v);
                if (WEIGHTED) {
                    C[(size_t)m * LDC + n_g] += wcol[m] * v;
                } else {
                    C[(size_t)m * LDC + n_g] = v;
                }
            }
        }
    }
}

// ---------------- top-2 over H=1024 logits, one wave32 per token ----------------
__global__ __launch_bounds__(256) void topk_kernel(
    const float* __restrict__ logits, float* __restrict__ wT,
    int Ntok, int Hdim, int E)
{
    const int wave = threadIdx.x >> 5;
    const int lane = threadIdx.x & 31;
    const int row  = blockIdx.x * 8 + wave;
    if (row >= Ntok) return;

    const float* p = logits + (size_t)row * Hdim;
    float v1 = -INFINITY, v2 = -INFINITY;
    int   i1 = 0x7fffffff, i2 = 0x7fffffff;

    for (int j = lane; j < Hdim; j += 32) {
        float v = p[j];
        if (v > v1 || (v == v1 && j < i1)) { v2 = v1; i2 = i1; v1 = v; i1 = j; }
        else if (v > v2 || (v == v2 && j < i2)) { v2 = v; i2 = j; }
    }
    // butterfly reduction over 32 lanes (disjoint groups each stage -> no dup merge)
    for (int off = 16; off >= 1; off >>= 1) {
        float ov1 = __shfl_xor(v1, off, 32);
        int   oi1 = __shfl_xor(i1, off, 32);
        float ov2 = __shfl_xor(v2, off, 32);
        int   oi2 = __shfl_xor(i2, off, 32);
        if (ov1 > v1 || (ov1 == v1 && oi1 < i1)) { v2 = v1; i2 = i1; v1 = ov1; i1 = oi1; }
        else if (ov1 > v2 || (ov1 == v2 && oi1 < i2)) { v2 = ov1; i2 = oi1; }
        if (ov2 > v1 || (ov2 == v1 && oi2 < i1)) { v2 = v1; i2 = i1; v1 = ov2; i1 = oi2; }
        else if (ov2 > v2 || (ov2 == v2 && oi2 < i2)) { v2 = ov2; i2 = oi2; }
    }
    if (lane == 0) {
        float inv = 1.f / (v1 + v2);          // same normalization (and hazards) as ref
        for (int e = 0; e < E; ++e) {
            float we = 0.f;
            if (i1 == e) we += v1 * inv;      // indices >= E silently dropped
            if (i2 == e) we += v2 * inv;
            wT[(size_t)e * Ntok + row] = we;
        }
    }
}

// per-(expert, BM-row-block) activity flags
__global__ __launch_bounds__(256) void mark_active(
    const float* __restrict__ wT, unsigned char* __restrict__ active,
    int Ntok, int nMB, int E)
{
    int idx = blockIdx.x * blockDim.x + threadIdx.x;
    if (idx >= E * nMB) return;
    int e = idx / nMB, mb = idx % nMB;
    const float* p = wT + (size_t)e * Ntok + (size_t)mb * BM;
    unsigned char f = 0;
    for (int r = 0; r < BM; ++r) f |= (p[r] != 0.f);
    active[idx] = f;
}

__global__ __launch_bounds__(256) void zero_f32(float* __restrict__ p, size_t n) {
    size_t i = (size_t)blockIdx.x * blockDim.x + threadIdx.x;
    size_t stride = (size_t)gridDim.x * blockDim.x;
    for (; i < n; i += stride) p[i] = 0.f;
}

extern "C" void kernel_launch(void* const* d_in, const int* in_sizes, int n_in,
                              void* d_out, int out_size, void* d_ws, size_t ws_size,
                              hipStream_t stream) {
    (void)in_sizes; (void)n_in; (void)out_size; (void)ws_size;
    const float* x       = (const float*)d_in[0];
    const float* gate_wi = (const float*)d_in[1];
    const float* gate_bi = (const float*)d_in[2];
    const float* gate_wo = (const float*)d_in[3];
    const float* gate_bo = (const float*)d_in[4];
    const float* exp_wi  = (const float*)d_in[5];
    const float* exp_bi  = (const float*)d_in[6];
    const float* exp_wo  = (const float*)d_in[7];
    const float* exp_bo  = (const float*)d_in[8];
    float* out = (float*)d_out;

    const int N = 2 * 2048, H = 1024, F = 4096, E = 8;
    const int nMB = N / BM;

    // workspace layout (~84 MB): h1[N,F] f32 | logits[N,H] f32 | wT[E,N] f32 | active
    char* ws = (char*)d_ws;
    float* h1     = (float*)ws;
    float* logits = (float*)(ws + (size_t)N * F * 4);
    float* wT     = (float*)(ws + (size_t)N * F * 4 + (size_t)N * H * 4);
    unsigned char* act = (unsigned char*)(ws + (size_t)N * F * 4 + (size_t)N * H * 4 + (size_t)E * N * 4);

    dim3 blk(256);

    // gate FFN: h1 = SELU(x@gate_wi + bi); logits = h1@gate_wo + bo
    gemm_bf16w<1, 0, 1024, 4096, 4096, 1024><<<dim3(F / BN, N / BM), blk, 0, stream>>>(
        x, gate_wi, gate_bi, h1, nullptr, nullptr);
    gemm_bf16w<0, 0, 4096, 1024, 1024, 4096><<<dim3(H / BN, N / BM), blk, 0, stream>>>(
        h1, gate_wo, gate_bo, logits, nullptr, nullptr);

    // routing weights + block activity map
    topk_kernel<<<N / 8, blk, 0, stream>>>(logits, wT, N, H, E);
    mark_active<<<(E * nMB + 255) / 256, blk, 0, stream>>>(wT, act, N, nMB, E);
    zero_f32<<<2048, blk, 0, stream>>>(out, (size_t)N * H);

    // experts: blocks with all-zero combine weight skipped entirely (contribute exact 0)
    for (int e = 0; e < E; ++e) {
        gemm_bf16w<1, 0, 1024, 4096, 4096, 1024><<<dim3(F / BN, N / BM), blk, 0, stream>>>(
            x, exp_wi + (size_t)e * H * F, exp_bi + (size_t)e * F,
            h1, nullptr, act + (size_t)e * nMB);
        gemm_bf16w<0, 1, 4096, 1024, 1024, 4096><<<dim3(H / BN, N / BM), blk, 0, stream>>>(
            h1, exp_wo + (size_t)e * F * H, exp_bo + (size_t)e * H,
            out, wT + (size_t)e * N, act + (size_t)e * nMB);
    }
}
